// Mirnet_63934883168995
// MI455X (gfx1250) — matrix operand order repaired and checked by reading, not compiled, not run on hardware
//
#include <hip/hip_runtime.h>
#include <hip/hip_bf16.h>
#include <math.h>

typedef __attribute__((ext_vector_type(16))) _Float16 v16h;
typedef __attribute__((ext_vector_type(8)))  _Float16 v8h;
typedef __attribute__((ext_vector_type(8)))  float    v8f;

// ---------------------------------------------------------------------------
// Workspace layout (float offsets)
// ---------------------------------------------------------------------------
static constexpr size_t BIG       = 32ull * 64 * 31 * 513;   // 32,569,344 floats
static constexpr size_t OFF_X     = 0;
static constexpr size_t OFF_Y     = BIG;
static constexpr size_t OFF_Z     = 2 * BIG;
static constexpr size_t OFF_S     = 3 * BIG;
static constexpr size_t OFF_MASK  = OFF_S;                   // 15,903
static constexpr size_t OFF_XM    = OFF_MASK + 16384;        // 508,896
static constexpr size_t OFF_FEATS = OFF_XM + 520192;         // 507,904
static constexpr size_t OFF_HS    = OFF_FEATS + 520192;      // 507,904
static constexpr size_t OFF_HF    = OFF_HS + 520192;         // 8192
static constexpr size_t OFF_CF    = OFF_HF + 8192;
static constexpr size_t OFF_HB    = OFF_CF + 8192;
static constexpr size_t OFF_CB    = OFF_HB + 8192;
static constexpr size_t OFF_GF    = OFF_CB + 8192;           // 32,768
static constexpr size_t OFF_GB    = OFF_GF + 32768;

__device__ __forceinline__ float sigmoidf_(float x) { return 1.0f / (1.0f + expf(-x)); }

// ---------------------------------------------------------------------------
// Filter mask:  mask[pos] = prod_k cos(fW[k,pos]*rng + fB[k,pos]),  rng = (pos%513)+1
// ---------------------------------------------------------------------------
__global__ void k_mask(const float* __restrict__ fW, const float* __restrict__ fB,
                       float* __restrict__ mask) {
    int pos = blockIdx.x * blockDim.x + threadIdx.x;
    if (pos >= 15903) return;
    float rng = (float)((pos % 513) + 1);
    float p = 1.0f;
    for (int k = 0; k < 1024; ++k) {
        size_t idx = (size_t)k * 15903 + pos;
        p *= cosf(fW[idx] * rng + fB[idx]);
    }
    mask[pos] = p;
}

__global__ void k_mask_apply(const float* __restrict__ x, const float* __restrict__ mask,
                             float* __restrict__ xm, int n) {
    int i = blockIdx.x * blockDim.x + threadIdx.x;
    if (i >= n) return;
    xm[i] = x[i] * mask[i % 15903];
}

// ---------------------------------------------------------------------------
// LDS-staged implicit-GEMM conv (NCHW, SAME) via V_WMMA_F32_16X16X32_F16.
// Block = 128 threads (4 waves). blockIdx.x -> 16 spatial columns,
// blockIdx.y -> 64 output channels (wave w owns rows w*16..w*16+15).
// Per 32-K chunk: cooperative stage of sA[64][32] (weights) and sB[16][32]
// (im2col patch, shared by all 4 waves), then vectorized LDS fragment loads.
// ---------------------------------------------------------------------------
__global__ void __launch_bounds__(128)
k_conv_wmma(const float* __restrict__ in, const float* __restrict__ wgt,
            float* __restrict__ out,
            int B, int Cin, int Cout, int H, int W, int KH, int KW) {
    __shared__ __align__(64) _Float16 sA[64 * 32];   // [m][k], k contiguous
    __shared__ __align__(64) _Float16 sB[16 * 32];   // [n][k], k contiguous

    const int tid   = threadIdx.x;          // 0..127
    const int wave  = tid >> 5;             // 0..3
    const int lane  = tid & 31;
    const int col   = lane & 15;
    const int hi    = lane >> 4;            // 0 or 1
    const int nBase = blockIdx.x * 16;
    const int coB0  = blockIdx.y * 64;      // Cout base of this block
    const int HW = H * W;
    const int N  = B * HW;
    const int KK = KH * KW;
    const int K  = Cin * KK;
    const int padh = KH >> 1, padw = KW >> 1;

    // --- staging roles ---
    // A: thread stages 16 halves of row mA = tid>>1, k offset (tid&1)*16
    const int mA = tid >> 1;
    const int kA = (tid & 1) * 16;
    const float* wrowA = wgt + (size_t)(coB0 + mA) * K;
    // B: thread stages 4 halves of column nS = tid/8, k offset (tid%8)*4
    const int nS = tid >> 3;
    const int kS = (tid & 7) * 4;
    const int sSt = nBase + nS;
    const bool sStOK = sSt < N;
    int bS = 0, hS = 0, wS = 0;
    if (sStOK) { bS = sSt / HW; int r = sSt % HW; hS = r / W; wS = r - hS * W; }

    // --- output column of this lane ---
    const int s = nBase + col;
    const bool sOK = s < N;
    int b = 0, h = 0, w = 0;
    if (sOK) { b = s / HW; int r = s % HW; h = r / W; w = r - h * W; }

    const int mRow = wave * 16 + col;       // local A row for this lane's fragment

    v8f acc = {};
    for (int k0 = 0; k0 < K; k0 += 32) {
        // ---- stage A (weights) ----
        #pragma unroll
        for (int j = 0; j < 16; ++j) {
            int kk = k0 + kA + j;
            sA[mA * 32 + kA + j] =
                (coB0 + mA < Cout && kk < K) ? (_Float16)wrowA[kk] : (_Float16)0.0f;
        }
        // prefetch next weight chunk for this row (global_prefetch_b8)
        if (k0 + 32 < K) __builtin_prefetch(wrowA + k0 + 32 + kA, 0, 1);

        // ---- stage B (im2col patch, shared by all 4 waves) ----
        #pragma unroll
        for (int j = 0; j < 4; ++j) {
            int kl = kS + j;
            int kk = k0 + kl;
            float val = 0.0f;
            if (sStOK && kk < K) {
                int ci = kk / KK;
                int r  = kk - ci * KK;
                int kh = r / KW;
                int kw = r - kh * KW;
                int ih = hS + kh - padh;
                int iw = wS + kw - padw;
                if (ih >= 0 && ih < H && iw >= 0 && iw < W)
                    val = in[(((size_t)bS * Cin + ci) * H + ih) * W + iw];
            }
            sB[nS * 32 + kl] = (_Float16)val;
        }
        __syncthreads();

        // ---- fragment loads from LDS (contiguous k order, same for A and B) ----
        // A lane (col,hi): elements i <-> k = 16*hi + i  (16 contiguous halves)
        union { v16h v; v8h h2[2]; } af;
        af.h2[0] = *(const v8h*)(sA + mRow * 32 + 16 * hi);
        af.h2[1] = *(const v8h*)(sA + mRow * 32 + 16 * hi + 8);
        // B lane (col,hi): elements i <-> k = 16*hi + i  (16 contiguous halves)
        v16h bf = *(const v16h*)(sB + col * 32 + 16 * hi);

        acc = __builtin_amdgcn_wmma_f32_16x16x32_f16(false, af.v, false, bf,
                                                     (short)0, acc, false, false);
        __syncthreads();
    }

    if (sOK) {
        #pragma unroll
        for (int v = 0; v < 8; ++v) {
            int co = coB0 + wave * 16 + v + (hi ? 8 : 0);
            if (co < Cout)
                out[(((size_t)b * Cout + co) * H + h) * W + w] = acc[v];
        }
    }
}

// ---------------------------------------------------------------------------
// Elementwise: BN (x*g/sqrt(1+eps)+b) + LeakyReLU(0.01)
// ---------------------------------------------------------------------------
__global__ void k_bn_lrelu(const float* __restrict__ in, const float* __restrict__ g,
                           const float* __restrict__ b, float* __restrict__ out,
                           int C, int HW, int n) {
    int i = blockIdx.x * blockDim.x + threadIdx.x;
    if (i >= n) return;
    int ch = (i / HW) % C;
    float v = in[i] * (g[ch] * rsqrtf(1.0f + 1e-5f)) + b[ch];
    out[i] = (v >= 0.0f) ? v : 0.01f * v;
}

// Max-pool width by 4 (VALID, stride 4)
__global__ void k_mp4(const float* __restrict__ in, float* __restrict__ out,
                      int n, int Wo, int Win) {
    int i = blockIdx.x * blockDim.x + threadIdx.x;
    if (i >= n) return;
    int wo = i % Wo;
    size_t base = (size_t)(i / Wo) * Win + wo * 4;
    float m = in[base];
    m = fmaxf(m, in[base + 1]);
    m = fmaxf(m, in[base + 2]);
    m = fmaxf(m, in[base + 3]);
    out[i] = m;
}

__global__ void k_add(const float* __restrict__ a, const float* __restrict__ b,
                      float* __restrict__ out, int n) {
    int i = blockIdx.x * blockDim.x + threadIdx.x;
    if (i < n) out[i] = a[i] + b[i];
}

__global__ void k_zero(float* __restrict__ p, int n) {
    int i = blockIdx.x * blockDim.x + threadIdx.x;
    if (i < n) p[i] = 0.0f;
}

// (32,256,31,2) -> feats (32,31,512)
__global__ void k_feats(const float* __restrict__ y, float* __restrict__ feats, int n) {
    int i = blockIdx.x * blockDim.x + threadIdx.x;
    if (i >= n) return;
    int k  = i & 511;
    int bt = i >> 9;
    int t  = bt % 31;
    int b  = bt / 31;
    int c  = k >> 1;
    int w  = k & 1;
    feats[i] = y[(((size_t)b * 256 + c) * 31 + t) * 2 + w];
}

// ---------------------------------------------------------------------------
// LSTM gates for one timestep: gates[b,j] = x_t[b,:]@wih[j,:] + h[b,:]@whh[j,:] + biases
// M=32 (2 tiles), N=1024 (64 tiles), K1=512, K2=256 — all WMMA.
// ---------------------------------------------------------------------------
__global__ void k_lstm_gates(const float* __restrict__ feats, const float* __restrict__ hprev,
                             const float* __restrict__ wih, const float* __restrict__ whh,
                             const float* __restrict__ bih, const float* __restrict__ bhh,
                             float* __restrict__ gates, int t) {
    const int lane = threadIdx.x;
    const int col  = lane & 15;
    const int hi   = lane >> 4;
    const int mBase = blockIdx.x * 16;   // batch rows (M = 32)
    const int nBase = blockIdx.y * 16;   // gate cols (N = 1024)

    const int bRow = mBase + col;
    const float* xrow = feats + ((size_t)bRow * 31 + t) * 512;
    const float* hrow = hprev + (size_t)bRow * 256;
    const float* wrow_ih = wih + (size_t)(nBase + col) * 512;
    const float* wrow_hh = whh + (size_t)(nBase + col) * 256;

    v8f acc = {};
    for (int k0 = 0; k0 < 512; k0 += 32) {
        v16h a, bf;
        #pragma unroll
        for (int v = 0; v < 8; ++v)
            #pragma unroll
            for (int e = 0; e < 2; ++e) {
                int ka = k0 + 2 * v + e + (hi ? 16 : 0);
                a[2 * v + e] = (_Float16)xrow[ka];
                int kb = k0 + 2 * v + e + (hi ? 16 : 0);
                bf[2 * v + e] = (_Float16)wrow_ih[kb];
            }
        acc = __builtin_amdgcn_wmma_f32_16x16x32_f16(false, a, false, bf,
                                                     (short)0, acc, false, false);
    }
    for (int k0 = 0; k0 < 256; k0 += 32) {
        v16h a, bf;
        #pragma unroll
        for (int v = 0; v < 8; ++v)
            #pragma unroll
            for (int e = 0; e < 2; ++e) {
                int ka = k0 + 2 * v + e + (hi ? 16 : 0);
                a[2 * v + e] = (_Float16)hrow[ka];
                int kb = k0 + 2 * v + e + (hi ? 16 : 0);
                bf[2 * v + e] = (_Float16)wrow_hh[kb];
            }
        acc = __builtin_amdgcn_wmma_f32_16x16x32_f16(false, a, false, bf,
                                                     (short)0, acc, false, false);
    }

    #pragma unroll
    for (int v = 0; v < 8; ++v) {
        int bb = mBase + v + (hi ? 8 : 0);
        int j  = nBase + col;
        gates[(size_t)bb * 1024 + j] = acc[v] + bih[j] + bhh[j];
    }
}

// LSTM pointwise update for one timestep; writes h into hs[t_out][b][choff..choff+255]
__global__ void k_lstm_point(const float* __restrict__ gates, float* __restrict__ h,
                             float* __restrict__ c, float* __restrict__ hs,
                             int t_out, int choff) {
    int i = blockIdx.x * blockDim.x + threadIdx.x;
    if (i >= 8192) return;
    int b = i >> 8, j = i & 255;
    const float* g = gates + (size_t)b * 1024;
    float ig = g[j], fg = g[256 + j], gg = g[512 + j], og = g[768 + j];
    float cc = sigmoidf_(fg) * c[i] + sigmoidf_(ig) * tanhf(gg);
    float hh = sigmoidf_(og) * tanhf(cc);
    c[i] = cc;
    h[i] = hh;
    hs[((size_t)t_out * 32 + b) * 512 + choff + j] = hh;
}

// ---------------------------------------------------------------------------
// Classifier: out[b*31+t, n] = hs[t,b,:] @ clsw[n,:] + clsb[n]   (M=992,N=722,K=512)
// ---------------------------------------------------------------------------
__global__ void k_classifier(const float* __restrict__ hs, const float* __restrict__ clsw,
                             const float* __restrict__ clsb, float* __restrict__ out) {
    const int lane = threadIdx.x;
    const int col  = lane & 15;
    const int hi   = lane >> 4;
    const int mBase = blockIdx.x * 16;   // M = 992 exactly
    const int nBase = blockIdx.y * 16;   // N = 722 (padded to 736)

    const int m  = mBase + col;
    const int bb = m / 31, tt = m % 31;
    const float* arow = hs + ((size_t)tt * 32 + bb) * 512;
    const int n = nBase + col;
    const bool nOK = n < 722;
    const float* brow = clsw + (size_t)(nOK ? n : 0) * 512;

    v8f acc = {};
    for (int k0 = 0; k0 < 512; k0 += 32) {
        v16h a, bf;
        #pragma unroll
        for (int v = 0; v < 8; ++v)
            #pragma unroll
            for (int e = 0; e < 2; ++e) {
                int ka = k0 + 2 * v + e + (hi ? 16 : 0);
                a[2 * v + e] = (_Float16)arow[ka];
                int kb = k0 + 2 * v + e + (hi ? 16 : 0);
                bf[2 * v + e] = nOK ? (_Float16)brow[kb] : (_Float16)0.0f;
            }
        acc = __builtin_amdgcn_wmma_f32_16x16x32_f16(false, a, false, bf,
                                                     (short)0, acc, false, false);
    }

    #pragma unroll
    for (int v = 0; v < 8; ++v) {
        int mo = mBase + v + (hi ? 8 : 0);
        int no = nBase + col;
        if (no < 722)
            out[(size_t)mo * 722 + no] = acc[v] + clsb[no];
    }
}

// ---------------------------------------------------------------------------
// Host helpers
// ---------------------------------------------------------------------------
static inline void conv(const float* in, const float* w, float* out,
                        int B, int Cin, int Cout, int H, int W, int KH, int KW,
                        hipStream_t s) {
    dim3 g((B * H * W + 15) / 16, Cout / 64);   // Cout is 64/128/192/256 here
    k_conv_wmma<<<g, dim3(128), 0, s>>>(in, w, out, B, Cin, Cout, H, W, KH, KW);
}
static inline void bnlrelu(const float* in, const float* g, const float* b, float* out,
                           int C, int HW, hipStream_t s) {
    int n = 32 * C * HW;
    k_bn_lrelu<<<(n + 255) / 256, 256, 0, s>>>(in, g, b, out, C, HW, n);
}

extern "C" void kernel_launch(void* const* d_in, const int* in_sizes, int n_in,
                              void* d_out, int out_size, void* d_ws, size_t ws_size,
                              hipStream_t stream) {
    (void)in_sizes; (void)n_in; (void)out_size; (void)ws_size;

    const float* x    = (const float*)d_in[0];
    const float* fW   = (const float*)d_in[1];
    const float* fB   = (const float*)d_in[2];
    const float* cbw1 = (const float*)d_in[3];
    const float* cbg  = (const float*)d_in[4];
    const float* cbb  = (const float*)d_in[5];
    const float* cbw2 = (const float*)d_in[6];
    const float* rp[3][7];   // pg, pb, wA, g, b, wB, s
    for (int r = 0; r < 3; ++r)
        for (int k = 0; k < 7; ++k)
            rp[r][k] = (const float*)d_in[7 + r * 7 + k];
    const float* pbg   = (const float*)d_in[28];
    const float* pbb   = (const float*)d_in[29];
    const float* wih_f = (const float*)d_in[30];
    const float* whh_f = (const float*)d_in[31];
    const float* bih_f = (const float*)d_in[32];
    const float* bhh_f = (const float*)d_in[33];
    const float* wih_b = (const float*)d_in[34];
    const float* whh_b = (const float*)d_in[35];
    const float* bih_b = (const float*)d_in[36];
    const float* bhh_b = (const float*)d_in[37];
    const float* clsw  = (const float*)d_in[38];
    const float* clsb  = (const float*)d_in[39];
    float* out = (float*)d_out;

    float* W  = (float*)d_ws;
    float* X  = W + OFF_X;
    float* Y  = W + OFF_Y;
    float* Z  = W + OFF_Z;
    float* MASK  = W + OFF_MASK;
    float* XM    = W + OFF_XM;
    float* FEATS = W + OFF_FEATS;
    float* HS    = W + OFF_HS;
    float* HF    = W + OFF_HF;
    float* CF    = W + OFF_CF;
    float* HB    = W + OFF_HB;
    float* CB    = W + OFF_CB;
    float* GF    = W + OFF_GF;
    float* GB    = W + OFF_GB;

    // ---- filter mask + apply ----
    k_mask<<<(15903 + 255) / 256, 256, 0, stream>>>(fW, fB, MASK);
    k_mask_apply<<<(508896 + 255) / 256, 256, 0, stream>>>(x, MASK, XM, 508896);

    // ---- conv block ----
    conv(XM, cbw1, X, 32, 1, 64, 31, 513, 3, 3, stream);
    bnlrelu(X, cbg, cbb, Y, 64, 15903, stream);
    conv(Y, cbw2, X, 32, 64, 64, 31, 513, 3, 3, stream);

    // ---- resblock 1 : 64ch W=513 -> 128ch W=128 ----
    bnlrelu(X, rp[0][0], rp[0][1], Y, 64, 15903, stream);
    k_mp4<<<(8142336 + 255) / 256, 256, 0, stream>>>(Y, Z, 8142336, 128, 513);
    conv(Z, rp[0][2], X, 32, 64, 128, 31, 128, 3, 3, stream);
    bnlrelu(X, rp[0][3], rp[0][4], Y, 128, 3968, stream);
    conv(Y, rp[0][5], X, 32, 128, 128, 31, 128, 3, 3, stream);
    conv(Z, rp[0][6], Z + 9000000, 32, 64, 128, 31, 128, 1, 1, stream);
    k_add<<<(16284672 + 255) / 256, 256, 0, stream>>>(X, Z + 9000000, Y, 16284672);

    // ---- resblock 2 : 128ch W=128 -> 192ch W=32 ----
    bnlrelu(Y, rp[1][0], rp[1][1], X, 128, 3968, stream);
    k_mp4<<<(4071168 + 255) / 256, 256, 0, stream>>>(X, Z, 4071168, 32, 128);
    conv(Z, rp[1][2], X, 32, 128, 192, 31, 32, 3, 3, stream);
    bnlrelu(X, rp[1][3], rp[1][4], Y, 192, 992, stream);
    conv(Y, rp[1][5], X, 32, 192, 192, 31, 32, 3, 3, stream);
    conv(Z, rp[1][6], Z + 5000000, 32, 128, 192, 31, 32, 1, 1, stream);
    k_add<<<(6106752 + 255) / 256, 256, 0, stream>>>(X, Z + 5000000, Y, 6106752);

    // ---- resblock 3 : 192ch W=32 -> 256ch W=8 ----
    bnlrelu(Y, rp[2][0], rp[2][1], X, 192, 992, stream);
    k_mp4<<<(1526688 + 255) / 256, 256, 0, stream>>>(X, Z, 1526688, 8, 32);
    conv(Z, rp[2][2], X, 32, 192, 256, 31, 8, 3, 3, stream);
    bnlrelu(X, rp[2][3], rp[2][4], Y, 256, 248, stream);
    conv(Y, rp[2][5], X, 32, 256, 256, 31, 8, 3, 3, stream);
    conv(Z, rp[2][6], Z + 2000000, 32, 192, 256, 31, 8, 1, 1, stream);
    k_add<<<(2035584 + 255) / 256, 256, 0, stream>>>(X, Z + 2000000, Y, 2035584);

    // ---- post block + feats ----
    bnlrelu(Y, pbg, pbb, X, 256, 248, stream);
    k_mp4<<<(507904 + 255) / 256, 256, 0, stream>>>(X, Z, 507904, 2, 8);
    k_feats<<<(507904 + 255) / 256, 256, 0, stream>>>(Z, FEATS, 507904);

    // ---- bidirectional LSTM ----
    k_zero<<<(32768 + 255) / 256, 256, 0, stream>>>(HF, 32768);  // HF,CF,HB,CB contiguous
    for (int s = 0; s < 31; ++s) {
        k_lstm_gates<<<dim3(2, 64), dim3(32), 0, stream>>>(
            FEATS, HF, wih_f, whh_f, bih_f, bhh_f, GF, s);
        k_lstm_point<<<32, 256, 0, stream>>>(GF, HF, CF, HS, s, 0);
        int tb = 30 - s;
        k_lstm_gates<<<dim3(2, 64), dim3(32), 0, stream>>>(
            FEATS, HB, wih_b, whh_b, bih_b, bhh_b, GB, tb);
        k_lstm_point<<<32, 256, 0, stream>>>(GB, HB, CB, HS, tb, 256);
    }

    // ---- classifier ----
    k_classifier<<<dim3(62, 46), dim3(32), 0, stream>>>(HS, clsw, clsb, out);
}
